// DRL4TSP_41875931136788
// MI455X (gfx1250) — matrix operand order repaired and checked by reading, not compiled, not run on hardware
//
#include <hip/hip_runtime.h>
#include <hip/hip_bf16.h>
#include <math.h>

// ---------------------------------------------------------------------------
// DRL4TSP pointer-network decoder for MI455X (gfx1250, wave32, WMMA).
// B=512, F=2, N=128, H=256. See analysis: all static/dyn-derived tensors are
// rank<=3 in F, so only h@W_hh.T and h@Wa_h.T are real GEMMs -> WMMA bf16.
// ---------------------------------------------------------------------------

#define B_   512
#define F_   2
#define N_   128
#define H_   256
#define H3_  768
#define CARLOAD 20.0f

typedef __attribute__((ext_vector_type(16))) __bf16 v16bf;
typedef __attribute__((ext_vector_type(8)))  float  v8f;

#define WMMA_BF16(A, Bm, C) \
  __builtin_amdgcn_wmma_f32_16x16x32_bf16(false, (A), false, (Bm), (short)0, (C), false, false)

__device__ __forceinline__ float sigmoidf_(float x) { return 1.0f / (1.0f + expf(-x)); }

// A fragment: 16x32 bf16 tile, row-major source (rows = M), per ISA 05_wmma
// 16-bit A layout: lanes0-15 hold K{0..7,16..23}, lanes16-31 hold K{8..15,24..31}.
__device__ __forceinline__ v16bf load_a_frag_lds(const __bf16* a, int lda, int k0, int lane) {
  const int lo = lane & 15, hi = lane >> 4;
  const __bf16* r = a + lo * lda + k0;
  v16bf v;
#pragma unroll
  for (int i = 0; i < 8; ++i) v[i] = r[hi * 8 + i];
#pragma unroll
  for (int i = 0; i < 8; ++i) v[8 + i] = r[16 + hi * 8 + i];
  return v;
}

// B fragment: 32x16 bf16 tile. Source is weight matrix W[col][k] row-major
// (output column j owns a contiguous K row). Per ISA: lanes0-15 K{0..7,16..23},
// lanes16-31 K{8..15,24..31}, column = lane&15.
__device__ __forceinline__ v16bf load_b_frag(const __bf16* w, int ldk, int col0, int k0, int lane) {
  const int lo = lane & 15, hi = lane >> 4;
  const __bf16* p = w + (size_t)(col0 + lo) * ldk + k0 + hi * 8;
  v16bf v;
#pragma unroll
  for (int i = 0; i < 16; ++i) v[i] = p[(i & 7) + (i >> 3) * 16];
  return v;
}

// ---------------------------------------------------------------------------
// One-time prep: fused small matrices (all F=2 rank collapses), bf16 weight
// copies for WMMA, h/idx init.
// ---------------------------------------------------------------------------
__global__ void k_prep(
    const float* __restrict__ Ws, const float* __restrict__ bs,
    const float* __restrict__ Wd, const float* __restrict__ bd,
    const float* __restrict__ Wdec, const float* __restrict__ bdec,
    const float* __restrict__ W_ih, const float* __restrict__ W_hh,
    const float* __restrict__ b_ih,
    const float* __restrict__ Wa, const float* __restrict__ Wp,
    const float* __restrict__ last_hh,
    float* __restrict__ Wc, float* __restrict__ bc,
    float* __restrict__ A0, float* __restrict__ A1,
    float* __restrict__ D1, float* __restrict__ Catt,
    float* __restrict__ P0, float* __restrict__ P1, float* __restrict__ Cptr,
    float* __restrict__ Pc0, float* __restrict__ Pc1, float* __restrict__ Pcb,
    __bf16* __restrict__ Whh_bf, __bf16* __restrict__ Wah_bf,
    float* __restrict__ h_f32, __bf16* __restrict__ h_bf, int* __restrict__ idx)
{
  const int gid = blockIdx.x * blockDim.x + threadIdx.x;
  const int nth = gridDim.x * blockDim.x;

  // Wc = W_ih @ Wdec (768x2), bc = b_ih + W_ih @ bdec
  for (int j = gid; j < H3_; j += nth) {
    float c0 = 0.f, c1 = 0.f, cb = 0.f;
    for (int k = 0; k < H_; ++k) {
      float w = W_ih[(size_t)j * H_ + k];
      c0 += w * Wdec[2 * k];
      c1 += w * Wdec[2 * k + 1];
      cb += w * bdec[k];
    }
    Wc[2 * j] = c0; Wc[2 * j + 1] = c1; bc[j] = b_ih[j] + cb;
  }

  // rank-collapsed attention / pointer coefficient vectors (length H)
  for (int h = gid; h < H_; h += nth) {
    float a0 = 0.f, a1 = 0.f, d1 = 0.f, ca = 0.f;
    float p0 = 0.f, p1 = 0.f, cp = 0.f, q0 = 0.f, q1 = 0.f, qb = 0.f;
    for (int k = 0; k < H_; ++k) {
      float was = Wa[(size_t)h * H3_ + k];          // Wa_s
      float wad = Wa[(size_t)h * H3_ + H_ + k];     // Wa_d
      a0 += was * Ws[2 * k];
      a1 += was * Ws[2 * k + 1];
      d1 += wad * Wd[2 * k + 1];
      ca += was * bs[k] + wad * (bd[k] + CARLOAD * Wd[2 * k]);  // dyn ch0 == 20
      float wps = Wp[(size_t)h * (2 * H_) + k];       // Wp_s
      float wpc = Wp[(size_t)h * (2 * H_) + H_ + k];  // Wp_c
      p0 += wps * Ws[2 * k];
      p1 += wps * Ws[2 * k + 1];
      cp += wps * bs[k];
      q0 += wpc * Ws[2 * k];
      q1 += wpc * Ws[2 * k + 1];
      qb += wpc * bs[k];
    }
    A0[h] = a0; A1[h] = a1; D1[h] = d1; Catt[h] = ca;
    P0[h] = p0; P1[h] = p1; Cptr[h] = cp;
    Pc0[h] = q0; Pc1[h] = q1; Pcb[h] = qb;
  }

  // bf16 copies for WMMA B operands
  for (int i = gid; i < H3_ * H_; i += nth) Whh_bf[i] = (__bf16)W_hh[i];
  for (int i = gid; i < H_ * H_; i += nth) {
    int h = i >> 8, k = i & 255;
    Wah_bf[i] = (__bf16)Wa[(size_t)h * H3_ + 2 * H_ + k];  // Wa_h
  }

  // hidden state init + idx init
  for (int i = gid; i < B_ * H_; i += nth) {
    float v = last_hh[i];
    h_f32[i] = v;
    h_bf[i] = (__bf16)v;
  }
  for (int i = gid; i < B_; i += nth) idx[i] = 0;
}

// ---------------------------------------------------------------------------
// Per-step kernel 1: GRU (gh = h @ W_hh.T via WMMA bf16) fused with gates and
// ha = h_new @ Wa_h.T (WMMA). 32 blocks x 256 threads (8 waves). Each wave
// owns column tiles {w, w+8, w+16, w+24, w+32, w+40} of the 768-wide gh, i.e.
// matching r/z/n tiles for the same h-range -> gates straight from the
// accumulators, no LDS staging of gh.
// ---------------------------------------------------------------------------
__global__ __launch_bounds__(256) void k_gru(
    const float* __restrict__ statc, const float* __restrict__ b_hh,
    const float* __restrict__ Wc, const float* __restrict__ bc,
    const __bf16* __restrict__ Whh_bf, const __bf16* __restrict__ Wah_bf,
    const int* __restrict__ idx,
    float* __restrict__ h_f32, __bf16* __restrict__ h_bf,
    float* __restrict__ ha)
{
  __shared__ __bf16 sH[16 * H_];   // h_prev (bf16) for 16 batch rows
  __shared__ __bf16 sHN[16 * H_];  // h_new  (bf16)
  __shared__ float sX0[16], sX1[16];

  const int tid  = threadIdx.x;
  const int wave = tid >> 5;
  const int lane = tid & 31;
  const int lo   = lane & 15;
  const int hi   = lane >> 4;
  const int b0   = blockIdx.x * 16;

  for (int i = tid; i < 16 * H_; i += 256)
    sH[i] = h_bf[(size_t)(b0 + (i >> 8)) * H_ + (i & 255)];
  if (tid < 16) {
    int b = b0 + tid;
    int id = idx[b];
    sX0[tid] = statc[(size_t)b * (F_ * N_) + id];        // static[b,0,idx]
    sX1[tid] = statc[(size_t)b * (F_ * N_) + N_ + id];   // static[b,1,idx]
  }
  __syncthreads();

  const v8f vz = {0.f, 0.f, 0.f, 0.f, 0.f, 0.f, 0.f, 0.f};
  v8f accR0 = vz, accR1 = vz, accZ0 = vz, accZ1 = vz, accN0 = vz, accN1 = vz;

#pragma unroll
  for (int kk = 0; kk < H_ / 32; ++kk) {
    const int k0 = kk * 32;
    v16bf a = load_a_frag_lds(sH, H_, k0, lane);
    v16bf bf0 = load_b_frag(Whh_bf, H_, (wave     ) * 16, k0, lane);
    v16bf bf1 = load_b_frag(Whh_bf, H_, (wave +  8) * 16, k0, lane);
    v16bf bf2 = load_b_frag(Whh_bf, H_, (wave + 16) * 16, k0, lane);
    v16bf bf3 = load_b_frag(Whh_bf, H_, (wave + 24) * 16, k0, lane);
    v16bf bf4 = load_b_frag(Whh_bf, H_, (wave + 32) * 16, k0, lane);
    v16bf bf5 = load_b_frag(Whh_bf, H_, (wave + 40) * 16, k0, lane);
    accR0 = WMMA_BF16(a, bf0, accR0);
    accR1 = WMMA_BF16(a, bf1, accR1);
    accZ0 = WMMA_BF16(a, bf2, accZ0);
    accZ1 = WMMA_BF16(a, bf3, accZ1);
    accN0 = WMMA_BF16(a, bf4, accN0);
    accN1 = WMMA_BF16(a, bf5, accN1);
  }

  // gates + h update, straight from WMMA accumulators
#pragma unroll
  for (int t = 0; t < 2; ++t) {
    const int th = wave + 8 * t;         // h-tile index 0..15
    const int hcol = th * 16 + lo;       // h column 0..255
    v8f aR = t ? accR1 : accR0;
    v8f aZ = t ? accZ1 : accZ0;
    v8f aN = t ? accN1 : accN0;
    const int jr = hcol, jz = hcol + H_, jn = hcol + 2 * H_;
    const float wcr0 = Wc[2 * jr], wcr1 = Wc[2 * jr + 1], bcr = bc[jr] + b_hh[jr];
    const float wcz0 = Wc[2 * jz], wcz1 = Wc[2 * jz + 1], bcz = bc[jz] + b_hh[jz];
    const float wcn0 = Wc[2 * jn], wcn1 = Wc[2 * jn + 1], bcn = bc[jn];
    const float bhn = b_hh[jn];
#pragma unroll
    for (int r = 0; r < 8; ++r) {
      const int m = r + 8 * hi;          // local batch row (C/D layout)
      const int b = b0 + m;
      const float x0 = sX0[m], x1 = sX1[m];
      float pre_r = wcr0 * x0 + wcr1 * x1 + bcr + aR[r];
      float pre_z = wcz0 * x0 + wcz1 * x1 + bcz + aZ[r];
      float gi_n  = wcn0 * x0 + wcn1 * x1 + bcn;
      float gh_n  = aN[r] + bhn;
      float rg = sigmoidf_(pre_r);
      float zg = sigmoidf_(pre_z);
      float ng = tanhf(gi_n + rg * gh_n);
      float hold = h_f32[(size_t)b * H_ + hcol];
      float hnew = (1.0f - zg) * ng + zg * hold;
      h_f32[(size_t)b * H_ + hcol] = hnew;
      __bf16 hb = (__bf16)hnew;
      h_bf[(size_t)b * H_ + hcol] = hb;
      sHN[m * H_ + hcol] = hb;
    }
  }
  __syncthreads();

  // ha = h_new @ Wa_h.T (16 x 256), each wave owns tiles w and w+8
  v8f accA0 = vz, accA1 = vz;
#pragma unroll
  for (int kk = 0; kk < H_ / 32; ++kk) {
    const int k0 = kk * 32;
    v16bf a = load_a_frag_lds(sHN, H_, k0, lane);
    v16bf bb0 = load_b_frag(Wah_bf, H_, (wave    ) * 16, k0, lane);
    v16bf bb1 = load_b_frag(Wah_bf, H_, (wave + 8) * 16, k0, lane);
    accA0 = WMMA_BF16(a, bb0, accA0);
    accA1 = WMMA_BF16(a, bb1, accA1);
  }
#pragma unroll
  for (int t = 0; t < 2; ++t) {
    v8f acc = t ? accA1 : accA0;
    const int col = (wave + 8 * t) * 16 + lo;
#pragma unroll
    for (int r = 0; r < 8; ++r) {
      const int m = r + 8 * hi;
      ha[(size_t)(b0 + m) * H_ + col] = acc[r];
    }
  }
}

// ---------------------------------------------------------------------------
// Per-step kernel 2: attention + pointer. One block per batch element,
// one thread per city (N=128). pre_attn / pre_ptr recomputed on the fly from
// rank-collapsed coefficients (3 FMA per element beats any memory traffic).
// ---------------------------------------------------------------------------
__device__ __forceinline__ float red_max128(float* buf, int t, float v) {
  buf[t] = v; __syncthreads();
#pragma unroll
  for (int o = 64; o > 0; o >>= 1) { if (t < o) buf[t] = fmaxf(buf[t], buf[t + o]); __syncthreads(); }
  float r = buf[0]; __syncthreads(); return r;
}
__device__ __forceinline__ float red_sum128(float* buf, int t, float v) {
  buf[t] = v; __syncthreads();
#pragma unroll
  for (int o = 64; o > 0; o >>= 1) { if (t < o) buf[t] += buf[t + o]; __syncthreads(); }
  float r = buf[0]; __syncthreads(); return r;
}
__device__ __forceinline__ int red_min128i(int* buf, int t, int v) {
  buf[t] = v; __syncthreads();
#pragma unroll
  for (int o = 64; o > 0; o >>= 1) { if (t < o) buf[t] = min(buf[t], buf[t + o]); __syncthreads(); }
  int r = buf[0]; __syncthreads(); return r;
}

__global__ __launch_bounds__(128) void k_attn(
    const float* __restrict__ statc, const float* __restrict__ dyn,
    const float* __restrict__ va, const float* __restrict__ vp,
    const float* __restrict__ A0, const float* __restrict__ A1,
    const float* __restrict__ D1, const float* __restrict__ Catt,
    const float* __restrict__ P0, const float* __restrict__ P1,
    const float* __restrict__ Cptr, const float* __restrict__ Pc0,
    const float* __restrict__ Pc1, const float* __restrict__ Pcb,
    const float* __restrict__ ha, int step,
    int* __restrict__ idx, float* __restrict__ out)
{
  __shared__ float sA0[H_], sA1[H_], sD1[H_], sC[H_], sVa[H_], sHa[H_];
  __shared__ float sP0[H_], sP1[H_], sCptr[H_], sVp[H_], sCp[H_];
  __shared__ float red[N_];
  __shared__ int redi[N_];

  const int t = threadIdx.x;   // city index n
  const int b = blockIdx.x;    // batch element

  for (int h = t; h < H_; h += N_) {
    sA0[h] = A0[h]; sA1[h] = A1[h]; sD1[h] = D1[h]; sC[h] = Catt[h];
    sVa[h] = va[h]; sHa[h] = ha[(size_t)b * H_ + h];
    sP0[h] = P0[h]; sP1[h] = P1[h]; sCptr[h] = Cptr[h]; sVp[h] = vp[h];
  }
  __syncthreads();

  const float s0 = statc[(size_t)b * (F_ * N_) + t];
  const float s1 = statc[(size_t)b * (F_ * N_) + N_ + t];
  const float d1 = dyn[(size_t)b * (F_ * N_) + N_ + t];

  // sweep 1: s[n] = sum_h va[h] * tanh(pre_attn[b,h,n] + ha[b,h])
  float s = 0.f;
#pragma unroll 4
  for (int h = 0; h < H_; ++h) {
    float u = tanhf(sA0[h] * s0 + sA1[h] * s1 + sD1[h] * d1 + sC[h] + sHa[h]);
    s += sVa[h] * u;
  }

  // softmax over n
  float smax = red_max128(red, t, s);
  float p = expf(s - smax);
  float psum = red_sum128(red, t, p);
  float attn = p / psum;

  // context collapses to two scalars (static_h is rank-2 + bias, sum attn = 1)
  float cs0 = red_sum128(red, t, attn * s0);
  float cs1 = red_sum128(red, t, attn * s1);

  for (int h = t; h < H_; h += N_)
    sCp[h] = Pc0[h] * cs0 + Pc1[h] * cs1 + Pcb[h];
  __syncthreads();

  // sweep 2: logits[n] = sum_h vp[h] * tanh(pre_ptr[b,h,n] + cp[b,h])
  float l = 0.f;
#pragma unroll 4
  for (int h = 0; h < H_; ++h) {
    float e = tanhf(sP0[h] * s0 + sP1[h] * s1 + sCptr[h] + sCp[h]);
    l += sVp[h] * e;
  }

  float lmax = red_max128(red, t, l);
  int arg = red_min128i(redi, t, (l == lmax) ? t : N_);    // first argmax
  float esum = red_sum128(red, t, expf(l - lmax));

  if (t == 0) {
    idx[b] = arg;
    out[(size_t)b * N_ + step] = (float)arg;                       // tour_idx.T
    out[(size_t)B_ * N_ + (size_t)b * N_ + step] = -logf(esum);    // tour_logp.T
  }
}

// ---------------------------------------------------------------------------
// Host launcher
// ---------------------------------------------------------------------------
extern "C" void kernel_launch(void* const* d_in, const int* in_sizes, int n_in,
                              void* d_out, int out_size, void* d_ws, size_t ws_size,
                              hipStream_t stream) {
  (void)in_sizes; (void)n_in; (void)out_size; (void)ws_size;

  const float* statc   = (const float*)d_in[0];
  const float* dynamc  = (const float*)d_in[1];
  const float* last_hh = (const float*)d_in[2];
  const float* Ws      = (const float*)d_in[3];
  const float* bs      = (const float*)d_in[4];
  const float* Wd      = (const float*)d_in[5];
  const float* bd      = (const float*)d_in[6];
  const float* Wdec    = (const float*)d_in[7];
  const float* bdec    = (const float*)d_in[8];
  const float* W_ih    = (const float*)d_in[9];
  const float* W_hh    = (const float*)d_in[10];
  const float* b_ih    = (const float*)d_in[11];
  const float* b_hh    = (const float*)d_in[12];
  const float* Wa      = (const float*)d_in[13];
  const float* va      = (const float*)d_in[14];
  const float* Wp      = (const float*)d_in[15];
  const float* vp      = (const float*)d_in[16];

  char* ws = (char*)d_ws;
  size_t off = 0;
  auto alloc = [&](size_t bytes) -> void* {
    void* p = ws + off;
    off = (off + bytes + 255) & ~(size_t)255;
    return p;
  };

  float*  Wc     = (float*)alloc(H3_ * 2 * sizeof(float));
  float*  bc     = (float*)alloc(H3_ * sizeof(float));
  float*  A0     = (float*)alloc(H_ * sizeof(float));
  float*  A1     = (float*)alloc(H_ * sizeof(float));
  float*  D1     = (float*)alloc(H_ * sizeof(float));
  float*  Catt   = (float*)alloc(H_ * sizeof(float));
  float*  P0     = (float*)alloc(H_ * sizeof(float));
  float*  P1     = (float*)alloc(H_ * sizeof(float));
  float*  Cptr   = (float*)alloc(H_ * sizeof(float));
  float*  Pc0    = (float*)alloc(H_ * sizeof(float));
  float*  Pc1    = (float*)alloc(H_ * sizeof(float));
  float*  Pcb    = (float*)alloc(H_ * sizeof(float));
  __bf16* Whh_bf = (__bf16*)alloc((size_t)H3_ * H_ * sizeof(__bf16));
  __bf16* Wah_bf = (__bf16*)alloc((size_t)H_ * H_ * sizeof(__bf16));
  float*  h_f32  = (float*)alloc((size_t)B_ * H_ * sizeof(float));
  __bf16* h_bf   = (__bf16*)alloc((size_t)B_ * H_ * sizeof(__bf16));
  float*  ha     = (float*)alloc((size_t)B_ * H_ * sizeof(float));
  int*    idx    = (int*)alloc(B_ * sizeof(int));

  float* out = (float*)d_out;

  k_prep<<<128, 256, 0, stream>>>(Ws, bs, Wd, bd, Wdec, bdec, W_ih, W_hh, b_ih,
                                  Wa, Wp, last_hh,
                                  Wc, bc, A0, A1, D1, Catt, P0, P1, Cptr,
                                  Pc0, Pc1, Pcb, Whh_bf, Wah_bf,
                                  h_f32, h_bf, idx);

  for (int step = 0; step < N_; ++step) {
    k_gru<<<B_ / 16, 256, 0, stream>>>(statc, b_hh, Wc, bc, Whh_bf, Wah_bf,
                                       idx, h_f32, h_bf, ha);
    k_attn<<<B_, N_, 0, stream>>>(statc, dynamc, va, vp,
                                  A0, A1, D1, Catt, P0, P1, Cptr,
                                  Pc0, Pc1, Pcb, ha, step, idx, out);
  }
}
